// NATLayer_1769526526661
// MI455X (gfx1250) — matrix operand order repaired and checked by reading, not compiled, not run on hardware
//
#include <hip/hip_runtime.h>
#include <hip/hip_bf16.h>

typedef __attribute__((ext_vector_type(16))) _Float16 v16h;
typedef __attribute__((ext_vector_type(8)))  _Float16 v8h;
typedef __attribute__((ext_vector_type(8)))  float    v8f;

#define NTOK   16384      // 4*64*64
#define CDIM   128
#define HID    512
#define HEADS  4
#define HD     32

// ---------------------------------------------------------------------------
// Weight convert + transpose: W[K][N] f32 -> Wt[N][K] f16  (coalesced read)
// ---------------------------------------------------------------------------
__global__ __launch_bounds__(256) void cvt_wt_kernel(
    const float* __restrict__ W, _Float16* __restrict__ Wt, int Kd, int Nout)
{
    int idx = blockIdx.x * 256 + threadIdx.x;
    if (idx >= Kd * Nout) return;
    int n = idx % Nout;
    int k = idx / Nout;
    Wt[(size_t)n * Kd + k] = (_Float16)W[idx];
}

// ---------------------------------------------------------------------------
// LayerNorm over 128 channels, one wave32 per token, f16 output
// ---------------------------------------------------------------------------
__global__ __launch_bounds__(256) void ln_kernel(
    const float4* __restrict__ x, const float* __restrict__ g,
    const float* __restrict__ b, _Float16* __restrict__ out, int ntok)
{
    int wid  = threadIdx.x >> 5;
    int lane = threadIdx.x & 31;
    int t = blockIdx.x * 8 + wid;
    if (t >= ntok) return;

    float4 xv = x[(size_t)t * 32 + lane];
    float s  = xv.x + xv.y + xv.z + xv.w;
    float s2 = xv.x*xv.x + xv.y*xv.y + xv.z*xv.z + xv.w*xv.w;
    #pragma unroll
    for (int off = 16; off; off >>= 1) {
        s  += __shfl_xor(s,  off);
        s2 += __shfl_xor(s2, off);
    }
    float m   = s  * (1.0f / 128.0f);
    float var = s2 * (1.0f / 128.0f) - m * m;
    float inv = rsqrtf(var + 1e-5f);

    int c = lane * 4;
    float xa[4] = { xv.x, xv.y, xv.z, xv.w };
    _Float16* op = out + (size_t)t * CDIM + c;
    #pragma unroll
    for (int i = 0; i < 4; i++)
        op[i] = (_Float16)((xa[i] - m) * inv * g[c + i] + b[c + i]);
}

// ---------------------------------------------------------------------------
// WMMA GEMM: C[M][Nout] = A[M][Kd](f16) * Bt[Nout][Kd](f16)^T + epilogue
// One wave computes a 16x64 strip: 4 accumulators share one A fragment
// (4 WMMAs per A load). K-loop is explicitly double-buffered: fragments for
// step i+1 are issued before the WMMAs of step i, forcing two register
// generations and letting WMMAs retire against in-flight loads.
// 8 waves/block -> 128x64 block tile. K-step 32, Kd compile-time.
// EPI 0: f32 out = acc + bias
// EPI 1: f32 out = acc + bias + resid
// EPI 2: f16 out = gelu(acc + bias)
// ---------------------------------------------------------------------------
template <int EPI, int KD>
__global__ __launch_bounds__(256) void gemm16_kernel(
    const _Float16* __restrict__ A, const _Float16* __restrict__ Bt,
    const float* __restrict__ bias, const float* __restrict__ resid,
    float* __restrict__ outF, _Float16* __restrict__ outH,
    int M, int Nout)
{
    constexpr int NT = 4;                 // 16-col tiles per wave
    constexpr int KSTEPS = KD / 32;
    int wid  = threadIdx.x >> 5;
    int lane = threadIdx.x & 31;
    int tileM  = (blockIdx.x * 8 + wid) * 16;
    int tileN0 = blockIdx.y * (16 * NT);
    if (tileM >= M) return;

    int g = lane >> 4;      // lane group 0/1
    int r = lane & 15;      // row-in-tile (A) / col-in-tile (B, C)

    const _Float16* arow = A + (size_t)(tileM + r) * KD;
    const _Float16* brow[NT];
    #pragma unroll
    for (int j = 0; j < NT; j++)
        brow[j] = Bt + (size_t)(tileN0 + j * 16 + r) * KD;

    v8f acc[NT];
    #pragma unroll
    for (int j = 0; j < NT; j++)
        acc[j] = (v8f){ 0.f, 0.f, 0.f, 0.f, 0.f, 0.f, 0.f, 0.f };

    union Frag { v16h v; v8h h[2]; };
    Frag fa[2];
    Frag fb[2][NT];

    // A 16x32 f16 layout: lane group g holds K in [16g, 16g+16)
    // B 32x16 f16 layout: lane group g holds K in [16g, 16g+16)
    auto load_step = [&](int k0, Frag& a, Frag (&bset)[NT]) {
        a.h[0] = *(const v8h*)(arow + k0 + 16 * g);
        a.h[1] = *(const v8h*)(arow + k0 + 16 * g + 8);
        #pragma unroll
        for (int j = 0; j < NT; j++) {
            bset[j].h[0] = *(const v8h*)(brow[j] + k0 + 16 * g);
            bset[j].h[1] = *(const v8h*)(brow[j] + k0 + 16 * g + 8);
        }
    };

    load_step(0, fa[0], fb[0]);
    #pragma unroll
    for (int ks = 0; ks < KSTEPS; ks++) {
        int cur = ks & 1;
        int nxt = cur ^ 1;
        if (ks + 1 < KSTEPS)
            load_step((ks + 1) * 32, fa[nxt], fb[nxt]);
        #pragma unroll
        for (int j = 0; j < NT; j++) {
            acc[j] = __builtin_amdgcn_wmma_f32_16x16x32_f16(
                false, fa[cur].v, false, fb[cur][j].v, (short)0, acc[j],
                false, false);
        }
    }

    #pragma unroll
    for (int j = 0; j < NT; j++) {
        int col = tileN0 + j * 16 + r;
        float bb = bias[col];
        #pragma unroll
        for (int i = 0; i < 8; i++) {
            int row = tileM + i + 8 * g;
            float v = acc[j][i] + bb;
            if constexpr (EPI == 1) {
                v += resid[(size_t)row * Nout + col];
                outF[(size_t)row * Nout + col] = v;
            } else if constexpr (EPI == 2) {
                float ge = 0.5f * v * (1.0f + erff(v * 0.70710678118f));
                outH[(size_t)row * Nout + col] = (_Float16)ge;
            } else {
                outF[(size_t)row * Nout + col] = v;
            }
        }
    }
}

// ---------------------------------------------------------------------------
// 7x7 neighborhood attention, one wave32 per (token, head).
// qkv: [NTOK][384] f32 (q|k|v).  rpb: [4][13][13].  out: [NTOK][128] f16
// ---------------------------------------------------------------------------
__global__ __launch_bounds__(256) void nat_attn_kernel(
    const float* __restrict__ qkv, const float* __restrict__ rpb,
    _Float16* __restrict__ out)
{
    __shared__ float sq[8][32];
    __shared__ float sp[8][64];

    int wid  = threadIdx.x >> 5;
    int lane = threadIdx.x & 31;
    int unit = blockIdx.x * 8 + wid;          // ((b*4 + h)*64 + y)*64 + x
    int x = unit & 63;
    int y = (unit >> 6) & 63;
    int h = (unit >> 12) & 3;
    int bb = unit >> 14;
    int t = ((bb * 64 + y) << 6) + x;

    // stage scaled q in LDS (lane = channel)
    float qv = qkv[(size_t)t * 384 + h * HD + lane] * 0.17677669529f; // 32^-0.5
    sq[wid][lane] = qv;

    int sy = y - 3; sy = sy < 0 ? 0 : (sy > 57 ? 57 : sy);
    int sx = x - 3; sx = sx < 0 ? 0 : (sx > 57 ? 57 : sx);

    // logits: lane owns neighbors lane and lane+32
    #pragma unroll
    for (int rep = 0; rep < 2; rep++) {
        int n = lane + rep * 32;
        float logit = -1e30f;
        if (n < 49) {
            int dy = n / 7, dx = n - dy * 7;
            int ky = sy + dy, kx = sx + dx;
            int kt = ((bb * 64 + ky) << 6) + kx;
            const float4* kp = (const float4*)(qkv + (size_t)kt * 384 + CDIM + h * HD);
            float dot = 0.f;
            #pragma unroll
            for (int c4 = 0; c4 < 8; c4++) {
                float4 kv = kp[c4];
                const float* qp = &sq[wid][c4 * 4];
                dot += qp[0]*kv.x + qp[1]*kv.y + qp[2]*kv.z + qp[3]*kv.w;
            }
            int ry = ky - y + 6, rx = kx - x + 6;
            logit = dot + rpb[(h * 13 + ry) * 13 + rx];
        }
        sp[wid][n] = logit;
    }

    // wave softmax over 49 logits
    float v0 = sp[wid][lane];
    float v1 = sp[wid][lane + 32];
    float mx = fmaxf(v0, v1);
    #pragma unroll
    for (int off = 16; off; off >>= 1) mx = fmaxf(mx, __shfl_xor(mx, off));
    float e0 = (v0 > -1e29f) ? __expf(v0 - mx) : 0.f;
    float e1 = (v1 > -1e29f) ? __expf(v1 - mx) : 0.f;
    float sum = e0 + e1;
    #pragma unroll
    for (int off = 16; off; off >>= 1) sum += __shfl_xor(sum, off);
    float inv = 1.0f / sum;
    sp[wid][lane]      = e0 * inv;
    sp[wid][lane + 32] = e1 * inv;

    // PV: lane = output channel
    float acc = 0.f;
    #pragma unroll
    for (int n = 0; n < 49; n++) {
        int dy = n / 7, dx = n - dy * 7;
        int ky = sy + dy, kx = sx + dx;
        int kt = ((bb * 64 + ky) << 6) + kx;
        acc += sp[wid][n] * qkv[(size_t)kt * 384 + 2 * CDIM + h * HD + lane];
    }
    out[(size_t)t * CDIM + h * HD + lane] = (_Float16)acc;
}

// ---------------------------------------------------------------------------
// Host-side orchestration
// ---------------------------------------------------------------------------
extern "C" void kernel_launch(void* const* d_in, const int* in_sizes, int n_in,
                              void* d_out, int out_size, void* d_ws, size_t ws_size,
                              hipStream_t stream) {
    const float* x      = (const float*)d_in[0];
    const float* ln1_g  = (const float*)d_in[1];
    const float* ln1_b  = (const float*)d_in[2];
    const float* qkv_w  = (const float*)d_in[3];
    const float* qkv_b  = (const float*)d_in[4];
    const float* rpb    = (const float*)d_in[5];
    const float* proj_w = (const float*)d_in[6];
    const float* proj_b = (const float*)d_in[7];
    const float* ln2_g  = (const float*)d_in[8];
    const float* ln2_b  = (const float*)d_in[9];
    const float* fc1_w  = (const float*)d_in[10];
    const float* fc1_b  = (const float*)d_in[11];
    const float* fc2_w  = (const float*)d_in[12];
    const float* fc2_b  = (const float*)d_in[13];
    float* out = (float*)d_out;

    char* ws = (char*)d_ws;
    _Float16* wqkv  = (_Float16*)(ws + 0);          //  384*128 f16
    _Float16* wproj = (_Float16*)(ws + 98304);      //  128*128 f16
    _Float16* wfc1  = (_Float16*)(ws + 131072);     //  512*128 f16
    _Float16* wfc2  = (_Float16*)(ws + 262144);     //  128*512 f16
    _Float16* xn16  = (_Float16*)(ws + 393216);     //  NTOK*128 f16  (reused for ln2 out)
    _Float16* att16 = (_Float16*)(ws + 4587520);    //  NTOK*128 f16
    float*    hbuf  = (float*)   (ws + 8781824);    //  NTOK*128 f32
    float*    qkvf  = (float*)   (ws + 17170432);   //  NTOK*384 f32
    _Float16* hid16 = (_Float16*)(ws + 17170432);   //  NTOK*512 f16 (aliases qkvf, dead by then)

    // 1) weights -> f16 transposed [Nout][K]
    cvt_wt_kernel<<<(128 * 384 + 255) / 256, 256, 0, stream>>>(qkv_w,  wqkv, 128, 384);
    cvt_wt_kernel<<<(128 * 128 + 255) / 256, 256, 0, stream>>>(proj_w, wproj, 128, 128);
    cvt_wt_kernel<<<(128 * 512 + 255) / 256, 256, 0, stream>>>(fc1_w,  wfc1, 128, 512);
    cvt_wt_kernel<<<(512 * 128 + 255) / 256, 256, 0, stream>>>(fc2_w,  wfc2, 512, 128);

    // 2) LN1 -> f16
    ln_kernel<<<NTOK / 8, 256, 0, stream>>>((const float4*)x, ln1_g, ln1_b, xn16, NTOK);

    // 3) QKV GEMM (f32 out for attention)  [16384 x 384, K=128]
    gemm16_kernel<0, 128><<<dim3(NTOK / 128, 384 / 64), 256, 0, stream>>>(
        xn16, wqkv, qkv_b, nullptr, qkvf, nullptr, NTOK, 384);

    // 4) neighborhood attention -> f16
    nat_attn_kernel<<<(NTOK * HEADS) / 8, 256, 0, stream>>>(qkvf, rpb, att16);

    // 5) proj GEMM + residual(x) -> h (f32)  [16384 x 128, K=128]
    gemm16_kernel<1, 128><<<dim3(NTOK / 128, 128 / 64), 256, 0, stream>>>(
        att16, wproj, proj_b, x, hbuf, nullptr, NTOK, 128);

    // 6) LN2 -> f16 (reuse xn16)
    ln_kernel<<<NTOK / 8, 256, 0, stream>>>((const float4*)hbuf, ln2_g, ln2_b, xn16, NTOK);

    // 7) FC1 + GELU -> f16  [16384 x 512, K=128]
    gemm16_kernel<2, 128><<<dim3(NTOK / 128, 512 / 64), 256, 0, stream>>>(
        xn16, wfc1, fc1_b, nullptr, nullptr, hid16, NTOK, 512);

    // 8) FC2 + residual(h) -> d_out (f32)  [16384 x 128, K=512]
    gemm16_kernel<1, 512><<<dim3(NTOK / 128, 128 / 64), 256, 0, stream>>>(
        hid16, wfc2, fc2_b, hbuf, out, nullptr, NTOK, 128);
}
